// EquiformerV2_Embedding_55516747268877
// MI455X (gfx1250) — matrix operand order repaired and checked by reading, not compiled, not run on hardware
//
#include <hip/hip_runtime.h>
#include <hip/hip_bf16.h>
#include <math.h>

typedef _Float16 f16;
typedef __attribute__((ext_vector_type(16))) _Float16 v16h;
typedef __attribute__((ext_vector_type(8)))  float    v8f;

#define N_EDGES_C 98304
#define N_ATOMS_C 16384
#define CDIM      128
#define NGAUSS    600
#define KPAD      608       // 600 padded to multiple of 32
#define NCOEF     49
#define D3COL     896       // 7*128
#define ZMAXV     90
#define CUTOFFV   5.0f
#define AVGDEG    23.395238876342773f
#define PI_F      3.14159265358979323846f

// ---------------- WMMA fragment loaders (16x16x32 f16, wave32) ----------------
__device__ __forceinline__ v16h ldA_lds(const f16* base, int stride, int k0, int lane) {
  int m = lane & 15, h = lane >> 4;
  v16h a;
  #pragma unroll
  for (int v = 0; v < 8; ++v) {
    int kb = (v < 4) ? (8*h + 2*v) : (16 + 8*h + 2*(v - 4));
    a[2*v]   = base[m*stride + k0 + kb];
    a[2*v+1] = base[m*stride + k0 + kb + 1];
  }
  return a;
}
__device__ __forceinline__ v16h ldB_pack(const f16* __restrict__ Wp, int numN, int kt, int nt, int lane) {
  return *(const v16h*)(Wp + (((size_t)(kt * numN + nt) * 32 + lane) << 4));
}
__device__ __forceinline__ float silu_f(float x) { return x / (1.f + __expf(-x)); }

// ---------------- precompute: f32 -> fragment-ready f16 B tiles ----------------
__global__ __launch_bounds__(256)
void k_pack_b(const float* __restrict__ W, f16* __restrict__ out,
              int numK, int numN, int ldw, int kValid) {
  int idx = blockIdx.x * 256 + threadIdx.x;
  if (idx >= numK * numN * 512) return;
  int j  = idx & 15;
  int L  = (idx >> 4) & 31;
  int t  = idx >> 9;
  int nt = t % numN;
  int kt = t / numN;
  int k  = kt * 32 + ((j < 8) ? (8 * (L >> 4) + j) : (16 + 8 * (L >> 4) + (j - 8)));
  int n  = nt * 16 + (L & 15);
  out[idx] = (k < kValid) ? (f16)W[(size_t)k * ldw + n] : (f16)0.f;
}

// per-atomic-number projections: src_emb @ W1[600:728], tgt_emb @ W1[728:856]
__global__ __launch_bounds__(256) void k_proj(const float* __restrict__ W1,
                                              const float* __restrict__ semb,
                                              const float* __restrict__ temb,
                                              float* __restrict__ srcP, float* __restrict__ tgtP) {
  int i = blockIdx.x * 256 + threadIdx.x;
  if (i >= 2 * ZMAXV * CDIM) return;
  int which = i / (ZMAXV * CDIM);
  int rem = i - which * ZMAXV * CDIM;
  int z = rem / CDIM, c = rem - z * CDIM;
  const float* emb = which ? temb : semb;
  int ro = which ? (NGAUSS + CDIM) : NGAUSS;
  float acc = 0.f;
  #pragma unroll 8
  for (int k = 0; k < CDIM; ++k)
    acc = fmaf(emb[z * CDIM + k], W1[(size_t)(ro + k) * CDIM + c], acc);
  (which ? tgtP : srcP)[rem] = acc;
}

// ---------------- CSR build: count -> scan -> fill ----------------
__global__ __launch_bounds__(256)
void k_count(const int* __restrict__ eidx, int* __restrict__ cnt) {
  int e = blockIdx.x * 256 + threadIdx.x;
  if (e < N_EDGES_C) atomicAdd(&cnt[eidx[N_EDGES_C + e]], 1);
}
__global__ __launch_bounds__(256)
void k_scan(const int* __restrict__ cnt, int* __restrict__ offs, int* __restrict__ cursor) {
  __shared__ int part[256];
  int t = threadIdx.x;
  int base = t * 64;                       // 16384 = 256 * 64
  int s = 0;
  for (int j = 0; j < 64; ++j) s += cnt[base + j];
  part[t] = s;
  __syncthreads();
  if (t == 0) {
    int run = 0;
    for (int i = 0; i < 256; ++i) { int tmp = part[i]; part[i] = run; run += tmp; }
  }
  __syncthreads();
  int run = part[t];
  for (int j = 0; j < 64; ++j) {
    offs[base + j] = run;
    cursor[base + j] = run;
    run += cnt[base + j];
  }
}
__global__ __launch_bounds__(256)
void k_fill(const int* __restrict__ eidx, int* __restrict__ cursor, int* __restrict__ list) {
  int e = blockIdx.x * 256 + threadIdx.x;
  if (e < N_EDGES_C) {
    int p = atomicAdd(&cursor[eidx[N_EDGES_C + e]], 1);
    list[p] = e;
  }
}

// ---------------- fused edge pipeline (32 edges / block) ----------------
// geometry -> Y_lm(->shW) -> GEMM1 (19x2 wmma) -> +bias,silu -> GEMM2 (4x2) -> silu
// -> GEMM3 per l-block (7 x 4x2) -> rad (f16) to workspace, coalesced via LDS staging
__global__ __launch_bounds__(256)
void k_edge_all(const float* __restrict__ pos, const int* __restrict__ az,
                const int* __restrict__ eidx,
                const float* __restrict__ b1, const float* __restrict__ b2,
                const float* __restrict__ b3,
                const f16* __restrict__ W1p, const f16* __restrict__ W2p,
                const f16* __restrict__ W3p,
                const float* __restrict__ srcP, const float* __restrict__ tgtP,
                float* __restrict__ shOut, f16* __restrict__ radOut) {
  // gauss tile, bias tile, and rad staging have disjoint lifetimes -> alias
  __shared__ __align__(16) unsigned char smemRaw[32 * KPAD * sizeof(f16)];   // 38,912 B
  __shared__ f16   h1[32][CDIM];         // 8,192 B
  __shared__ float distS[32];
  __shared__ int   zsS[32], ztS[32];

  f16   (*gaussS)[KPAD]  = (f16 (*)[KPAD])smemRaw;
  float (*addC)[CDIM]    = (float (*)[CDIM])smemRaw;
  f16   (*radStage)[CDIM]= (f16 (*)[CDIM])smemRaw;

  int tid = threadIdx.x;
  int lane = tid & 31, w = tid >> 5;
  int e0 = blockIdx.x * 32;

  float ct = 0.f, st = 0.f, phi = 0.f;
  if (tid < 32) {
    int e = e0 + tid;
    int s = eidx[e], t = eidx[N_EDGES_C + e];
    zsS[tid] = az[s];
    ztS[tid] = az[t];
    float vx = pos[3*s+0] - pos[3*t+0];
    float vy = pos[3*s+1] - pos[3*t+1];
    float vz = pos[3*s+2] - pos[3*t+2];
    float d  = sqrtf(vx*vx + vy*vy + vz*vz + 1e-12f);
    distS[tid] = d;
    float iz = vz / d;
    ct = fminf(fmaxf(iz, -1.f), 1.f);
    st = sqrtf(fmaxf(1.f - ct*ct, 1e-12f));
    phi = atan2f(vy / d, vx / d);
  }
  __syncthreads();

  const float delta = CUTOFFV / (NGAUSS - 1);
  const float coeff = -0.5f / ((2.f * delta) * (2.f * delta));
  for (int i = tid; i < 32 * KPAD; i += 256) {
    int el = i / KPAD, j = i - el * KPAD;
    float g = 0.f;
    if (j < NGAUSS) {
      float dd = distS[el] - (float)j * delta;
      g = __expf(coeff * dd * dd);
    }
    gaussS[el][j] = (f16)g;
  }

  // spherical harmonics Y_lm -> global (wave 0: 32 edges <-> 32 lanes)
  if (tid < 32) {
    float P[28];
    #define PIDX(l,m) ((l)*((l)+1)/2 + (m))
    P[PIDX(0,0)] = 1.f;
    for (int m = 1; m <= 6; ++m) P[PIDX(m,m)] = -(2.f*m - 1.f) * st * P[PIDX(m-1,m-1)];
    for (int m = 0; m <= 5; ++m) P[PIDX(m+1,m)] = (2.f*m + 1.f) * ct * P[PIDX(m,m)];
    for (int m = 0; m <= 6; ++m)
      for (int l = m + 2; l <= 6; ++l)
        P[PIDX(l,m)] = ((2.f*l - 1.f) * ct * P[PIDX(l-1,m)] - (l + m - 1.f) * P[PIDX(l-2,m)]) / (float)(l - m);
    float* sh = shOut + (size_t)(e0 + tid) * NCOEF;
    for (int l = 0; l <= 6; ++l) {
      for (int m = 0; m <= l; ++m) {
        float ratio = 1.f;
        for (int i = l - m + 1; i <= l + m; ++i) ratio /= (float)i;
        float Nlm = sqrtf((2.f*l + 1.f) / (4.f * PI_F) * ratio);
        if (m == 0) {
          sh[l*l + l] = Nlm * P[PIDX(l,0)];
        } else {
          float base = 1.41421356237309515f * Nlm * P[PIDX(l,m)];
          sh[l*l + l + m] = base * cosf((float)m * phi);
          sh[l*l + l - m] = base * sinf((float)m * phi);
        }
      }
    }
    #undef PIDX
  }
  __syncthreads();

  int n0 = 16 * w;
  int n = lane & 15, h = lane >> 4;
  // GEMM1: [32 x 608] x [608 x 128]; dual M-tile chains
  v8f acc0 = {}, acc1 = {};
  #pragma unroll
  for (int kt = 0; kt < KPAD / 32; ++kt) {
    v16h b  = ldB_pack(W1p, 8, kt, w, lane);
    v16h a0 = ldA_lds(&gaussS[0][0],  KPAD, kt * 32, lane);
    v16h a1 = ldA_lds(&gaussS[16][0], KPAD, kt * 32, lane);
    acc0 = __builtin_amdgcn_wmma_f32_16x16x32_f16(false, a0, false, b, (short)0, acc0, false, false);
    acc1 = __builtin_amdgcn_wmma_f32_16x16x32_f16(false, a1, false, b, (short)0, acc1, false, false);
  }
  __syncthreads();                               // gauss dead -> becomes addC
  for (int i = tid; i < 32 * CDIM; i += 256) {
    int el = i >> 7, c = i & 127;
    addC[el][c] = b1[c] + srcP[zsS[el] * CDIM + c] + tgtP[ztS[el] * CDIM + c];
  }
  __syncthreads();
  #pragma unroll
  for (int r = 0; r < 8; ++r) {
    h1[r + 8*h][n0 + n]      = (f16)silu_f(acc0[r] + addC[r + 8*h][n0 + n]);
    h1[16 + r + 8*h][n0 + n] = (f16)silu_f(acc1[r] + addC[16 + r + 8*h][n0 + n]);
  }
  __syncthreads();
  // GEMM2: [32 x 128] x [128 x 128]
  v8f d0, d1;
  float bb2 = b2[n0 + n];
  #pragma unroll
  for (int r = 0; r < 8; ++r) { d0[r] = bb2; d1[r] = bb2; }
  #pragma unroll
  for (int kt = 0; kt < CDIM / 32; ++kt) {
    v16h b  = ldB_pack(W2p, 8, kt, w, lane);
    v16h a0 = ldA_lds(&h1[0][0],  CDIM, kt * 32, lane);
    v16h a1 = ldA_lds(&h1[16][0], CDIM, kt * 32, lane);
    d0 = __builtin_amdgcn_wmma_f32_16x16x32_f16(false, a0, false, b, (short)0, d0, false, false);
    d1 = __builtin_amdgcn_wmma_f32_16x16x32_f16(false, a1, false, b, (short)0, d1, false, false);
  }
  __syncthreads();                               // all waves done reading h1
  #pragma unroll
  for (int r = 0; r < 8; ++r) {
    h1[r + 8*h][n0 + n]      = (f16)silu_f(d0[r]);
    h1[16 + r + 8*h][n0 + n] = (f16)silu_f(d1[r]);
  }
  __syncthreads();                               // h1 read-only from here

  // GEMM3 per l-block: rad = h @ W3[:, lb*128 : +128] + b3 -> f16, staged to global
  for (int lb = 0; lb <= 6; ++lb) {
    v8f r0, r1;
    float bb3 = b3[lb * CDIM + n0 + n];
    #pragma unroll
    for (int r = 0; r < 8; ++r) { r0[r] = bb3; r1[r] = bb3; }
    #pragma unroll
    for (int kt = 0; kt < CDIM / 32; ++kt) {
      v16h b  = ldB_pack(W3p, D3COL / 16, kt, lb * 8 + w, lane);
      v16h a0 = ldA_lds(&h1[0][0],  CDIM, kt * 32, lane);
      v16h a1 = ldA_lds(&h1[16][0], CDIM, kt * 32, lane);
      r0 = __builtin_amdgcn_wmma_f32_16x16x32_f16(false, a0, false, b, (short)0, r0, false, false);
      r1 = __builtin_amdgcn_wmma_f32_16x16x32_f16(false, a1, false, b, (short)0, r1, false, false);
    }
    __syncthreads();                             // previous staging copy done
    #pragma unroll
    for (int r = 0; r < 8; ++r) {
      radStage[r + 8*h][n0 + n]      = (f16)r0[r];
      radStage[16 + r + 8*h][n0 + n] = (f16)r1[r];
    }
    __syncthreads();
    // coalesced copy: 32 rows x 256B; rad layout [e][7][128] f16
    for (int c = tid; c < 512; c += 256) {
      int el = c >> 4, q = c & 15;
      ((uint4*)radOut)[((size_t)(e0 + el) * 7 + lb) * 16 + q] =
          ((const uint4*)&radStage[0][0])[c];
    }
  }
}

// ---------------- per-atom gather + RMS-norm (no atomics, single output write) ----------------
__global__ __launch_bounds__(256)
void k_gather_norm(const float* __restrict__ shW, const f16* __restrict__ radW,
                   const int* __restrict__ offs, const int* __restrict__ cnt,
                   const int* __restrict__ list, const int* __restrict__ az,
                   const float* __restrict__ sphere, const float* __restrict__ gamma,
                   float* __restrict__ out) {
  __shared__ float red[256];
  int a = blockIdx.x, t = threadIdx.x;
  int beg = offs[a], num = cnt[a];
  // fixed per-thread element mapping: idx = t + 256*i ; k = idx/128 ; l = floor(sqrt(k))
  int kI[25], rO[25];
  #pragma unroll
  for (int i = 0; i < 25; ++i) {
    int idx = t + 256 * i;
    if (idx < NCOEF * CDIM) {
      int k = idx >> 7;
      int l = (int)sqrtf((float)k + 0.25f);
      kI[i] = k;
      rO[i] = l * CDIM + (idx & 127);
    } else { kI[i] = 0; rO[i] = 0; }
  }
  float v[25];
  #pragma unroll
  for (int i = 0; i < 25; ++i) v[i] = 0.f;
  for (int j = 0; j < num; ++j) {
    int e = list[beg + j];
    const float* sh = shW + (size_t)e * NCOEF;
    const f16*  rad = radW + (size_t)e * D3COL;
    #pragma unroll
    for (int i = 0; i < 25; ++i) {
      int idx = t + 256 * i;
      if (idx < NCOEF * CDIM)
        v[i] += sh[kI[i]] * (float)rad[rO[i]];
    }
  }
  int z = az[a];
  float ss = 0.f;
  #pragma unroll
  for (int i = 0; i < 25; ++i) {
    int idx = t + 256 * i;
    if (idx < NCOEF * CDIM) {
      float x = v[i] * (1.f / AVGDEG);
      if (idx < CDIM) x += sphere[(size_t)z * CDIM + idx];
      v[i] = x;
      ss += x * x;
    }
  }
  red[t] = ss;
  __syncthreads();
  for (int sN = 128; sN > 0; sN >>= 1) {
    if (t < sN) red[t] += red[t + sN];
    __syncthreads();
  }
  float inv = 1.f / sqrtf(red[0] / (float)(NCOEF * CDIM) + 1e-6f);
  size_t base = (size_t)a * (NCOEF * CDIM);
  #pragma unroll
  for (int i = 0; i < 25; ++i) {
    int idx = t + 256 * i;
    if (idx < NCOEF * CDIM)
      out[base + idx] = v[i] * inv * gamma[idx & 127];
  }
}

extern "C" void kernel_launch(void* const* d_in, const int* in_sizes, int n_in,
                              void* d_out, int out_size, void* d_ws, size_t ws_size,
                              hipStream_t stream) {
  (void)in_sizes; (void)n_in; (void)ws_size; (void)out_size;
  const float* pos    = (const float*)d_in[0];
  const int*   az     = (const int*)d_in[1];
  const int*   eidx   = (const int*)d_in[2];
  const float* sphere = (const float*)d_in[3];
  const float* semb   = (const float*)d_in[4];
  const float* temb   = (const float*)d_in[5];
  const float* W1     = (const float*)d_in[6];
  const float* b1     = (const float*)d_in[7];
  const float* W2     = (const float*)d_in[8];
  const float* b2     = (const float*)d_in[9];
  const float* W3     = (const float*)d_in[10];
  const float* b3     = (const float*)d_in[11];
  const float* gamma  = (const float*)d_in[12];

  // workspace layout (~197 MB, all offsets 256B aligned)
  char* w = (char*)d_ws;
  f16*   W1p    = (f16*)  (w + 0);          // 155,648 B
  f16*   W2p    = (f16*)  (w + 155648);     //  32,768 B
  f16*   W3p    = (f16*)  (w + 188416);     // 229,376 B
  float* srcP   = (float*)(w + 417792);     //  46,080 B
  float* tgtP   = (float*)(w + 463872);     //  46,080 B
  int*   cnt    = (int*)  (w + 509952);     //  65,536 B
  int*   cursor = (int*)  (w + 575488);     //  65,536 B
  int*   offs   = (int*)  (w + 641024);     //  65,536 B
  int*   list   = (int*)  (w + 706560);     // 393,216 B
  float* shW    = (float*)(w + 1099776);    // E*49 f32  = 19,267,584 B
  f16*   radW   = (f16*)  (w + 20367360);   // E*896 f16 = 176,160,768 B

  // zero only the CSR counters (no 411MB output memset needed anymore)
  hipMemsetAsync(cnt, 0, N_ATOMS_C * sizeof(int), stream);

  k_pack_b<<<(19 * 8 * 512 + 255) / 256, 256, 0, stream>>>(W1, W1p, 19, 8, CDIM, NGAUSS);
  k_pack_b<<<(4 * 8 * 512 + 255) / 256, 256, 0, stream>>>(W2, W2p, 4, 8, CDIM, CDIM);
  k_pack_b<<<(4 * 56 * 512 + 255) / 256, 256, 0, stream>>>(W3, W3p, 4, 56, D3COL, CDIM);
  k_proj<<<(2 * ZMAXV * CDIM + 255) / 256, 256, 0, stream>>>(W1, semb, temb, srcP, tgtP);

  // CSR by target atom
  k_count<<<N_EDGES_C / 256, 256, 0, stream>>>(eidx, cnt);
  k_scan<<<1, 256, 0, stream>>>(cnt, offs, cursor);
  k_fill<<<N_EDGES_C / 256, 256, 0, stream>>>(eidx, cursor, list);

  // edge pipeline (all GEMMs, WMMA) -> sh, rad
  k_edge_all<<<N_EDGES_C / 32, 256, 0, stream>>>(pos, az, eidx, b1, b2, b3,
                                                 W1p, W2p, W3p, srcP, tgtP, shW, radW);

  // per-atom gather + fused RMS-norm -> final output
  k_gather_norm<<<N_ATOMS_C, 256, 0, stream>>>(shW, radW, offs, cnt, list, az,
                                               sphere, gamma, (float*)d_out);
}
